// LatentActionGen_49761491092176
// MI455X (gfx1250) — matrix operand order repaired and checked by reading, not compiled, not run on hardware
//
#include <hip/hip_runtime.h>
#include <hip/hip_bf16.h>

// MI455X / gfx1250, wave32, WMMA bf16 path.
// Pipeline: x = relu(s0@W0 + s1@W1 + b0 + b1); flat = x@Wf + bf; VQ(K=30,E=4).

typedef __attribute__((ext_vector_type(16))) __bf16 v16bf;
typedef __attribute__((ext_vector_type(8)))  __bf16 v8bf;
typedef __attribute__((ext_vector_type(8)))  float  v8f;
typedef __attribute__((ext_vector_type(4)))  float  f4;

#define N_TOT  32768
#define D_IN   4096
#define H_DIM  128
#define E_DIM  4
#define K_CB   30
#define M_TILE 64
#define LDA    40       // padded bf16 stride for staged A tiles (80B: 16B-aligned, bank-spread)
#define LDX    132      // padded float stride for x tile (528B)
#define ABUF_B (M_TILE * LDA * 2)   // bytes per A buffer (5120)
#define NSTEP  (D_IN / 32)

// -------------------------------------------------------------------------
// Kernel 0: zero global histogram in workspace
// -------------------------------------------------------------------------
__global__ void lag_zero(int* __restrict__ ghist) {
    if (threadIdx.x < K_CB) ghist[threadIdx.x] = 0;
}

// -------------------------------------------------------------------------
// Kernel 1: repack W0/W1 (fp32 [4096][128]) into bf16 B-fragment-major tiles.
// Tile (m, kb, cb) = 32 lanes x 16 bf16 (32B/lane, 1KB/tile).
// Lane l: col = cb*16 + (l&15); element e: K = kb*32 + (l>=16)*16 + e
// -> contiguous k order, the same order the A fragments are read from LDS.
// -------------------------------------------------------------------------
__global__ void lag_repack(const float* __restrict__ W0,
                           const float* __restrict__ W1,
                           __bf16* __restrict__ outB) {
    int gid  = blockIdx.x * blockDim.x + threadIdx.x;   // 0..65535
    int lane = gid & 31;
    int tile = gid >> 5;                                // m*1024 + kb*8 + cb
    int cbk  = tile & 7;
    int kb   = (tile >> 3) & 127;
    int m    = tile >> 10;
    const float* W = m ? W1 : W0;
    int col   = (cbk << 4) + (lane & 15);
    int kbase = (kb << 5) + ((lane >> 4) << 4);
    __bf16* dst = outB + (size_t)tile * 512 + lane * 16;
#pragma unroll
    for (int e = 0; e < 16; ++e)
        dst[e] = (__bf16)W[(size_t)(kbase + e) * H_DIM + col];
}

// -------------------------------------------------------------------------
// helpers
// -------------------------------------------------------------------------
__device__ inline v8bf cvt8(f4 a, f4 b) {
    v8bf r;
    r[0] = (__bf16)a[0]; r[1] = (__bf16)a[1]; r[2] = (__bf16)a[2]; r[3] = (__bf16)a[3];
    r[4] = (__bf16)b[0]; r[5] = (__bf16)b[1]; r[6] = (__bf16)b[2]; r[7] = (__bf16)b[3];
    return r;
}

// -------------------------------------------------------------------------
// Kernel 2: fused MLP + VQ. 512 blocks x 256 threads (8 waves).
// Block owns 64 rows; wave w owns H-columns [w*16, w*16+16).
// Double-buffered LDS A staging (1 barrier / K-step); s0/s1 loaded
// non-temporal (streaming) so weights stay L2-resident.
// -------------------------------------------------------------------------
__global__ __launch_bounds__(256)
void lag_main(const float* __restrict__ s0, const float* __restrict__ s1,
              const float* __restrict__ b0, const float* __restrict__ b1,
              const float* __restrict__ bf_, const float* __restrict__ cbk,
              const float* __restrict__ Wf,
              const __bf16* __restrict__ wsB,
              float* __restrict__ out_z, float* __restrict__ out_loss,
              float* __restrict__ out_idx, int* __restrict__ ghist) {
    // Union region: during K-loop -> 4 A-staging buffers (2 x double-buffered);
    // after K-loop -> fp32 x tile [64][LDX]. 33792 bytes.
    __shared__ __align__(16) unsigned char smem_u[M_TILE * LDX * 4];
    __shared__ __align__(16) float  sWf[E_DIM * H_DIM];
    __shared__ __align__(16) float  sCB[K_CB * E_DIM];
    __shared__ float sFlat[M_TILE * E_DIM];
    __shared__ int   sHist[K_CB];

    const int t    = threadIdx.x;
    const int lane = t & 31;
    const int w    = t >> 5;        // wave id 0..7 -> H column block
    const int hi   = lane >> 4;     // K-half selector
    const int ln16 = lane & 15;
    const size_t row0 = (size_t)blockIdx.x * M_TILE;

    // one-time staging of Wf^T, codebook, histogram
    if (t < H_DIM) {
#pragma unroll
        for (int e = 0; e < E_DIM; ++e) sWf[e * H_DIM + t] = Wf[t * E_DIM + e];
    }
    if (t < K_CB * E_DIM) sCB[t] = cbk[t];
    if (t < K_CB) sHist[t] = 0;

    const int   c     = (w << 4) + ln16;       // this lane's output column
    const float biasc = b0[c] + b1[c];

    v8f acc[4];
#pragma unroll
    for (int m = 0; m < 4; ++m)
#pragma unroll
        for (int i = 0; i < 8; ++i) acc[m][i] = 0.0f;

    // staging role: thread t loads row (t>>2), 8 floats at k-quarter (t&3)
    const int srow  = t >> 2;
    const int sq    = t & 3;
    const int stoff = srow * LDA + sq * 8;     // bf16-element offset inside a buffer
    const float* g0 = s0 + (row0 + srow) * D_IN + sq * 8;
    const float* g1 = s1 + (row0 + srow) * D_IN + sq * 8;

    const v16bf* Bb = (const v16bf*)wsB;  // 32 v16bf per tile

    // ---- prologue: stage K-step 0 into buffer 0 ----
    {
        f4 a0 = __builtin_nontemporal_load((const f4*)(g0));
        f4 a1 = __builtin_nontemporal_load((const f4*)(g0 + 4));
        f4 a2 = __builtin_nontemporal_load((const f4*)(g1));
        f4 a3 = __builtin_nontemporal_load((const f4*)(g1 + 4));
        *(v8bf*)((__bf16*)(smem_u) + stoff)              = cvt8(a0, a1);
        *(v8bf*)((__bf16*)(smem_u + 2 * ABUF_B) + stoff) = cvt8(a2, a3);
        g0 += 32; g1 += 32;
    }
    __syncthreads();

#pragma unroll 1
    for (int kb = 0; kb < NSTEP; ++kb) {
        const int p = kb & 1;

        // ---- issue next slab's streaming loads early (latency under WMMAs) ----
        f4 a0, a1, a2, a3;
        const bool have_next = (kb + 1 < NSTEP);
        if (have_next) {
            a0 = __builtin_nontemporal_load((const f4*)(g0));
            a1 = __builtin_nontemporal_load((const f4*)(g0 + 4));
            a2 = __builtin_nontemporal_load((const f4*)(g1));
            a3 = __builtin_nontemporal_load((const f4*)(g1 + 4));
            g0 += 32; g1 += 32;
            if (kb + 2 < NSTEP) {           // global_prefetch_b8 two slabs ahead
                __builtin_prefetch(g0, 0, 0);
                __builtin_prefetch(g1, 0, 0);
            }
        }

        // ---- B fragments (pre-repacked, fully coalesced 32B/lane, L2-resident) ----
        v16bf bw0 = Bb[(size_t)(kb * 8 + w) * 32 + lane];
        v16bf bw1 = Bb[(size_t)(kb * 8 + w) * 32 + lane + 32768];

        // current-parity A buffers (computed, not a pointer table -> no static init)
        const __bf16* Ab0 = (const __bf16*)(smem_u + (size_t)p * ABUF_B);
        const __bf16* Ab1 = (const __bf16*)(smem_u + 2 * ABUF_B + (size_t)p * ABUF_B);

        // ---- 4 M-subtiles x (s0@W0 + s1@W1) chained into one accumulator ----
#pragma unroll
        for (int m = 0; m < 4; ++m) {
            const __bf16* pa0 = Ab0 + (m * 16 + ln16) * LDA + hi * 16;
            v8bf lo0 = *(const v8bf*)pa0;
            v8bf hh0 = *(const v8bf*)(pa0 + 8);
            v16bf af0 = __builtin_shufflevector(lo0, hh0,
                0,1,2,3,4,5,6,7,8,9,10,11,12,13,14,15);
            acc[m] = __builtin_amdgcn_wmma_f32_16x16x32_bf16(
                false, af0, false, bw0, (short)0, acc[m], false, false);

            const __bf16* pa1 = Ab1 + (m * 16 + ln16) * LDA + hi * 16;
            v8bf lo1 = *(const v8bf*)pa1;
            v8bf hh1 = *(const v8bf*)(pa1 + 8);
            v16bf af1 = __builtin_shufflevector(lo1, hh1,
                0,1,2,3,4,5,6,7,8,9,10,11,12,13,14,15);
            acc[m] = __builtin_amdgcn_wmma_f32_16x16x32_bf16(
                false, af1, false, bw1, (short)0, acc[m], false, false);
        }

        // ---- convert + stage next slab into the other-parity buffer ----
        if (have_next) {
            __bf16* Sb0 = (__bf16*)(smem_u + (size_t)(1 - p) * ABUF_B);
            __bf16* Sb1 = (__bf16*)(smem_u + 2 * ABUF_B + (size_t)(1 - p) * ABUF_B);
            *(v8bf*)(Sb0 + stoff) = cvt8(a0, a1);
            *(v8bf*)(Sb1 + stoff) = cvt8(a2, a3);
        }
        __syncthreads();   // single barrier per K-step
    }

    // ---- epilogue: bias + relu, scatter x tile to LDS (reuses A region) ----
    float* sX = (float*)smem_u;
#pragma unroll
    for (int m = 0; m < 4; ++m)
#pragma unroll
        for (int v = 0; v < 8; ++v) {
            float x = acc[m][v] + biasc;
            x = fmaxf(x, 0.0f);
            sX[(m * 16 + hi * 8 + v) * LDX + c] = x;
        }
    __syncthreads();

    // ---- flat = x @ Wf + bf : thread (r = t>>2, e = t&3) ----
    {
        const int r = t >> 2;
        const int e = t & 3;
        const float* xr = &sX[r * LDX];
        const float* we = &sWf[e * H_DIM];
        float dot = bf_[e];
#pragma unroll 8
        for (int h = 0; h < H_DIM; h += 4) {
            f4 xv = *(const f4*)(xr + h);
            f4 wv = *(const f4*)(we + h);
            dot += xv[0] * wv[0] + xv[1] * wv[1] + xv[2] * wv[2] + xv[3] * wv[3];
        }
        sFlat[r * E_DIM + e] = dot;
    }
    __syncthreads();

    // ---- VQ: argmin over 30 codes, z / loss / idx, LDS histogram ----
    if (t < M_TILE) {
        float f0 = sFlat[t * 4 + 0], f1 = sFlat[t * 4 + 1];
        float f2 = sFlat[t * 4 + 2], f3 = sFlat[t * 4 + 3];
        int bi = 0; float best = 3.4e38f;
#pragma unroll 1
        for (int k = 0; k < K_CB; ++k) {
            float d0 = f0 - sCB[k * 4 + 0];
            float d1 = f1 - sCB[k * 4 + 1];
            float d2 = f2 - sCB[k * 4 + 2];
            float d3 = f3 - sCB[k * 4 + 3];
            float d = d0 * d0 + d1 * d1 + d2 * d2 + d3 * d3;
            if (d < best) { best = d; bi = k; }
        }
        float q0 = sCB[bi * 4 + 0], q1 = sCB[bi * 4 + 1];
        float q2 = sCB[bi * 4 + 2], q3 = sCB[bi * 4 + 3];
        size_t gr = row0 + t;
        f4 zq; zq[0] = q0; zq[1] = q1; zq[2] = q2; zq[3] = q3;
        *(f4*)(&out_z[gr * 4]) = zq;           // z == quantized (STE)
        float l0 = q0 - f0, l1 = q1 - f1, l2 = q2 - f2, l3 = q3 - f3;
        // loss = q_latent + e_latent = 2 * mean((q-f)^2) = 0.5 * sum((q-f)^2)
        out_loss[gr] = 0.5f * (l0 * l0 + l1 * l1 + l2 * l2 + l3 * l3);
        out_idx[gr] = (float)bi;
        atomicAdd(&sHist[bi], 1);
    }
    __syncthreads();
    if (t < K_CB) atomicAdd(&ghist[t], sHist[t]);
}

// -------------------------------------------------------------------------
// Kernel 3: perplexity from code-usage histogram
// -------------------------------------------------------------------------
__global__ void lag_ppl(const int* __restrict__ ghist, float* __restrict__ out) {
    if (threadIdx.x == 0) {
        float s = 0.0f;
        for (int k = 0; k < K_CB; ++k) {
            float p = (float)ghist[k] * (1.0f / (float)N_TOT);
            s += p * logf(p + 1e-10f);
        }
        out[0] = expf(-s);
    }
}

// -------------------------------------------------------------------------
extern "C" void kernel_launch(void* const* d_in, const int* in_sizes, int n_in,
                              void* d_out, int out_size, void* d_ws, size_t ws_size,
                              hipStream_t stream) {
    const float* s0  = (const float*)d_in[0];
    const float* s1  = (const float*)d_in[1];
    const float* W0  = (const float*)d_in[2];
    const float* b0  = (const float*)d_in[3];
    const float* W1  = (const float*)d_in[4];
    const float* b1  = (const float*)d_in[5];
    const float* Wf  = (const float*)d_in[6];
    const float* bf_ = (const float*)d_in[7];
    const float* cbk = (const float*)d_in[8];

    float* out      = (float*)d_out;
    float* out_z    = out;                               // [N, 4]
    float* out_loss = out + (size_t)N_TOT * E_DIM;       // [N]
    float* out_ppl  = out_loss + N_TOT;                  // [1]
    float* out_idx  = out_ppl + 1;                       // [N] (as float)

    int*    ghist = (int*)d_ws;                          // 30 ints
    __bf16* wsB   = (__bf16*)((char*)d_ws + 512);        // 2 MB repacked weights

    lag_zero  <<<1,   32,  0, stream>>>(ghist);
    lag_repack<<<256, 256, 0, stream>>>(W0, W1, wsB);
    lag_main  <<<N_TOT / M_TILE, 256, 0, stream>>>(
        s0, s1, b0, b1, bf_, cbk, Wf, wsB,
        out_z, out_loss, out_idx, ghist);
    lag_ppl   <<<1,   32,  0, stream>>>(ghist, out_ppl);
}
